// CharRNN_51092930953888
// MI455X (gfx1250) — matrix operand order repaired and checked by reading, not compiled, not run on hardware
//
#include <hip/hip_runtime.h>
#include <hip/hip_bf16.h>
#include <math.h>

#define VOCAB 32000
#define BATCH 32
#define SEQ   128
#define HID   512
#define LAYERS 2

typedef __attribute__((ext_vector_type(16))) __bf16        v16bf;
typedef __attribute__((ext_vector_type(8)))  float         v8f;
typedef __attribute__((ext_vector_type(4)))  unsigned int  v4u;

union BFrag {
  v16bf          v;
  v4u            q[2];
  unsigned short u[16];
};

__device__ inline unsigned short f32_to_bf16(float f) {
  unsigned int u = __float_as_uint(f);
  unsigned int r = 0x7FFFu + ((u >> 16) & 1u);   // round-to-nearest-even
  u += r;
  return (unsigned short)(u >> 16);
}

__device__ inline float sigmoidf_(float x) { return 1.0f / (1.0f + __expf(-x)); }

// ---------------------------------------------------------------------------
// Wave 16x16 tile accumulate, vectorized fragments, direct global loads.
//   A  : [16 x K] bf16 row-major, stride lda  (tile origin)
//   Bt : B stored transposed: [16 cols x K] bf16 row-major, stride ldbt
// ---------------------------------------------------------------------------
__device__ inline v8f wmma_accum_g(v8f acc,
                                   const unsigned short* __restrict__ A, int lda,
                                   const unsigned short* __restrict__ Bt, int ldbt,
                                   int K, int lane) {
  const int half = (lane >> 4) << 3;   // 0 or 8
  const int bk   = (lane >> 4) << 4;   // 0 or 16
  const unsigned short* Ap = A + (size_t)(lane & 15) * lda;
  const unsigned short* Bp = Bt + (size_t)(lane & 15) * ldbt;
  for (int k0 = 0; k0 < K; k0 += 32) {
    BFrag a, b;
    a.q[0] = *(const v4u*)(Ap + k0 + half);
    a.q[1] = *(const v4u*)(Ap + k0 + 16 + half);
    b.q[0] = *(const v4u*)(Bp + k0 + half);
    b.q[1] = *(const v4u*)(Bp + k0 + 16 + half);
    acc = __builtin_amdgcn_wmma_f32_16x16x32_bf16(false, a.v, false, b.v,
                                                  (short)0, acc, false, false);
  }
  return acc;
}

// ---------------------------------------------------------------------------
// Setup kernels
// ---------------------------------------------------------------------------
__global__ void zero_state_kernel(unsigned short* __restrict__ hbf,
                                  float* __restrict__ c, int n) {
  int i = blockIdx.x * blockDim.x + threadIdx.x;
  if (i < n) { hbf[i] = 0; c[i] = 0.0f; }
}

__global__ void embed_kernel(const int* __restrict__ ids,
                             const float* __restrict__ emb,
                             unsigned short* __restrict__ x) {
  size_t n = (size_t)BATCH * SEQ * HID;
  for (size_t i = blockIdx.x * (size_t)blockDim.x + threadIdx.x; i < n;
       i += (size_t)gridDim.x * blockDim.x) {
    size_t bt = i / HID;
    int   hh = (int)(i % HID);
    int   tok = ids[bt];
    x[i] = f32_to_bf16(emb[(size_t)tok * HID + hh]);
  }
}

// Tiled transpose + f32->bf16: src [K][N] f32  ->  dst [N][K] bf16
__global__ __launch_bounds__(256)
void transpose_bf16_kernel(const float* __restrict__ src,
                           unsigned short* __restrict__ dst, int K, int N) {
  __shared__ float tile[32][33];
  const int n0 = blockIdx.x * 32;
  const int k0 = blockIdx.y * 32;
  const int tx = threadIdx.x;   // 0..31
  const int ty = threadIdx.y;   // 0..7
#pragma unroll
  for (int i = 0; i < 4; ++i) {
    int k = k0 + ty * 4 + i;
    tile[ty * 4 + i][tx] = src[(size_t)k * N + n0 + tx];
  }
  __syncthreads();
#pragma unroll
  for (int i = 0; i < 4; ++i) {
    int n = n0 + ty * 4 + i;
    dst[(size_t)n * K + k0 + tx] = f32_to_bf16(tile[tx][ty * 4 + i]);
  }
}

// ---------------------------------------------------------------------------
// LSTM gates:  gates[32, 2048] = [x | h] @ W + b  (two K-segments of 512)
// W stored transposed: W_t [4H n][2H k].  256 wave-tiles, 128 blocks x 64.
// ---------------------------------------------------------------------------
__global__ __launch_bounds__(64)
void lstm_gates_kernel(const unsigned short* __restrict__ Ain, int lda_in,
                       const unsigned short* __restrict__ Ah,   // [32,512]
                       const unsigned short* __restrict__ Wt,   // [2048,1024]
                       const float* __restrict__ bias,          // [2048]
                       float* __restrict__ gates) {             // [32,2048]
  const int wave = (blockIdx.x * 64 + threadIdx.x) >> 5;  // 0..255
  const int lane = threadIdx.x & 31;
  const int mt = wave >> 7;     // 0..1
  const int nt = wave & 127;    // 0..127
  const int row0 = mt * 16;
  const int col0 = nt * 16;
  v8f acc = {0.f,0.f,0.f,0.f,0.f,0.f,0.f,0.f};
  // input segment (k = 0..511)
  acc = wmma_accum_g(acc, Ain + (size_t)row0 * lda_in, lda_in,
                     Wt + (size_t)col0 * (2 * HID), 2 * HID, HID, lane);
  // recurrent segment (k = 512..1023)
  acc = wmma_accum_g(acc, Ah + (size_t)row0 * HID, HID,
                     Wt + (size_t)col0 * (2 * HID) + HID, 2 * HID, HID, lane);
  const int crow = row0 + ((lane >> 4) << 3);
  const int ccol = col0 + (lane & 15);
  const float bv = bias[ccol];
#pragma unroll
  for (int r = 0; r < 8; ++r)
    gates[(size_t)(crow + r) * (4 * HID) + ccol] = acc[r] + bv;
}

// ---------------------------------------------------------------------------
// LSTM pointwise cell update (TF gate order i,j,f,o; forget_bias = 0)
// ---------------------------------------------------------------------------
__global__ void lstm_point_kernel(const float* __restrict__ gates,
                                  float* __restrict__ c,
                                  unsigned short* __restrict__ hbf,
                                  unsigned short* __restrict__ hout,
                                  int is_last, int t) {
  int i = blockIdx.x * blockDim.x + threadIdx.x;
  if (i >= BATCH * HID) return;
  int b = i / HID, hh = i % HID;
  const float* g = gates + (size_t)b * 4 * HID;
  float gi = g[hh];
  float gj = g[HID + hh];
  float gf = g[2 * HID + hh];
  float go = g[3 * HID + hh];
  float nc = c[i] * sigmoidf_(gf) + sigmoidf_(gi) * tanhf(gj);
  float nh = tanhf(nc) * sigmoidf_(go);
  c[i] = nc;
  unsigned short hb = f32_to_bf16(nh);
  hbf[i] = hb;
  if (is_last)
    hout[(size_t)b * SEQ * HID + (size_t)t * HID + hh] = hb;
}

// ---------------------------------------------------------------------------
// Logits GEMM: out[4096,32000] = Hact[4096,512] @ Wsm[512,32000] + sb
// Wsm stored transposed: Wsmt [32000][512].
// Block: 256 thr (8 waves) -> 128(M) x 128(N).  Double-buffered LDS:
// next K-chunk's global_load_b128 issue BEFORE computing the current chunk,
// so L2 latency overlaps the 8 WMMAs + ds_load_b128 fragment reads.
// ---------------------------------------------------------------------------
__global__ __launch_bounds__(256)
void logits_kernel(const unsigned short* __restrict__ Hact,  // [4096,512]
                   const unsigned short* __restrict__ Wsmt,  // [32000,512]
                   const float* __restrict__ sb,
                   float* __restrict__ out) {
  __shared__ __align__(16) unsigned short sA[2][128 * 32];  // 2 x 8 KB
  __shared__ __align__(16) unsigned short sB[2][128 * 32];  // 2 x 8 KB
  const int tidx = threadIdx.x;
  const int lane = tidx & 31;
  const int wav  = tidx >> 5;                // 0..7 -> N sub-tile
  const int mrow0 = blockIdx.y * 128;
  const int ncol0 = blockIdx.x * 128;

  const int half = (lane >> 4) << 3;
  const int bk   = (lane >> 4) << 4;
  const int arow = lane & 15;
  const int bcol = (wav << 4) + (lane & 15); // 0..127 within block

  // each thread stages v4u elements tidx and tidx+256 (of 512) per matrix
  const int r0 = tidx >> 2, q8 = (tidx & 3) * 8;
  const int r1 = r0 + 64;
  const unsigned short* aSrc0 = Hact + (size_t)(mrow0 + r0) * HID + q8;
  const unsigned short* aSrc1 = Hact + (size_t)(mrow0 + r1) * HID + q8;
  const unsigned short* bSrc0 = Wsmt + (size_t)(ncol0 + r0) * HID + q8;
  const unsigned short* bSrc1 = Wsmt + (size_t)(ncol0 + r1) * HID + q8;

  v8f acc[8];
#pragma unroll
  for (int m = 0; m < 8; ++m) acc[m] = (v8f){0.f,0.f,0.f,0.f,0.f,0.f,0.f,0.f};

  // prologue: stage chunk 0
  v4u ra0 = *(const v4u*)aSrc0;
  v4u ra1 = *(const v4u*)aSrc1;
  v4u rb0 = *(const v4u*)bSrc0;
  v4u rb1 = *(const v4u*)bSrc1;
  ((v4u*)sA[0])[tidx]       = ra0;
  ((v4u*)sA[0])[tidx + 256] = ra1;
  ((v4u*)sB[0])[tidx]       = rb0;
  ((v4u*)sB[0])[tidx + 256] = rb1;
  __syncthreads();

  const int NCHUNK = HID / 32;   // 16
  for (int c = 0; c < NCHUNK; ++c) {
    const int cur = c & 1;
    if (c + 1 < NCHUNK) {        // issue next chunk's loads now
      const int k = (c + 1) * 32;
      ra0 = *(const v4u*)(aSrc0 + k);
      ra1 = *(const v4u*)(aSrc1 + k);
      rb0 = *(const v4u*)(bSrc0 + k);
      rb1 = *(const v4u*)(bSrc1 + k);
    }

    BFrag b;
    b.q[0] = *(const v4u*)(sB[cur] + bcol * 32 + half);
    b.q[1] = *(const v4u*)(sB[cur] + bcol * 32 + 16 + half);
#pragma unroll
    for (int m = 0; m < 8; ++m) {
      BFrag a;
      const int abase = ((m << 4) + arow) * 32;
      a.q[0] = *(const v4u*)(sA[cur] + abase + half);
      a.q[1] = *(const v4u*)(sA[cur] + abase + 16 + half);
      acc[m] = __builtin_amdgcn_wmma_f32_16x16x32_bf16(false, a.v, false, b.v,
                                                       (short)0, acc[m],
                                                       false, false);
    }

    if (c + 1 < NCHUNK) {        // fill the other buffer
      __syncthreads();
      ((v4u*)sA[cur ^ 1])[tidx]       = ra0;
      ((v4u*)sA[cur ^ 1])[tidx + 256] = ra1;
      ((v4u*)sB[cur ^ 1])[tidx]       = rb0;
      ((v4u*)sB[cur ^ 1])[tidx + 256] = rb1;
      __syncthreads();
    }
  }

  const int ccol = ncol0 + bcol;
  const float bv = sb[ccol];
#pragma unroll
  for (int m = 0; m < 8; ++m) {
    const int crow = mrow0 + (m << 4) + ((lane >> 4) << 3);
#pragma unroll
    for (int r = 0; r < 8; ++r)
      out[(size_t)(crow + r) * VOCAB + ccol] = acc[m][r] + bv;
  }
}

// ---------------------------------------------------------------------------
// One-pass online log-softmax cross-entropy per row, then deterministic mean
// ---------------------------------------------------------------------------
__global__ __launch_bounds__(256)
void rowloss_kernel(const float* __restrict__ logits,
                    const int* __restrict__ targets,
                    float* __restrict__ rowloss) {
  __shared__ float rm[256], rs[256];
  const int row = blockIdx.x;
  const float* x = logits + (size_t)row * VOCAB;
  float m = -3.4e38f, s = 0.f;
  for (int i = threadIdx.x; i < VOCAB; i += 256) {
    float v = x[i];
    if (v > m) { s = s * __expf(m - v) + 1.0f; m = v; }
    else        s += __expf(v - m);
  }
  rm[threadIdx.x] = m; rs[threadIdx.x] = s; __syncthreads();
  for (int st = 128; st > 0; st >>= 1) {
    if (threadIdx.x < st) {
      float m1 = rm[threadIdx.x], s1 = rs[threadIdx.x];
      float m2 = rm[threadIdx.x + st], s2 = rs[threadIdx.x + st];
      float M = fmaxf(m1, m2);
      rm[threadIdx.x] = M;
      rs[threadIdx.x] = s1 * __expf(m1 - M) + s2 * __expf(m2 - M);
    }
    __syncthreads();
  }
  if (threadIdx.x == 0)
    rowloss[row] = rm[0] + __logf(rs[0]) - x[targets[row]];
}

__global__ void cost_kernel(const float* __restrict__ rowloss,
                            float* __restrict__ cost) {
  __shared__ float red[256];
  float s = 0.f;
  for (int i = threadIdx.x; i < BATCH * SEQ; i += 256) s += rowloss[i];
  red[threadIdx.x] = s; __syncthreads();
  for (int st = 128; st > 0; st >>= 1) {
    if (threadIdx.x < st) red[threadIdx.x] += red[threadIdx.x + st];
    __syncthreads();
  }
  if (threadIdx.x == 0) *cost = red[0] / (float)(BATCH * SEQ);
}

// ---------------------------------------------------------------------------
extern "C" void kernel_launch(void* const* d_in, const int* in_sizes, int n_in,
                              void* d_out, int out_size, void* d_ws, size_t ws_size,
                              hipStream_t stream) {
  const int*   input_data = (const int*)d_in[0];
  const int*   targets    = (const int*)d_in[1];
  const float* embedding  = (const float*)d_in[2];
  const float* lstm_w     = (const float*)d_in[3];
  const float* lstm_b     = (const float*)d_in[4];
  const float* softmax_w  = (const float*)d_in[5];
  const float* softmax_b  = (const float*)d_in[6];
  float* out = (float*)d_out;

  char* ws = (char*)d_ws;
  size_t off = 0;
  auto alloc = [&](size_t bytes) -> void* {
    void* p = ws + off;
    off = (off + bytes + 255) & ~(size_t)255;
    return p;
  };
  unsigned short* x_bf16 = (unsigned short*)alloc((size_t)BATCH * SEQ * HID * 2);
  unsigned short* W_t    = (unsigned short*)alloc((size_t)LAYERS * 2 * HID * 4 * HID * 2);
  unsigned short* Wsm_t  = (unsigned short*)alloc((size_t)HID * VOCAB * 2);
  unsigned short* h_bf16 = (unsigned short*)alloc((size_t)LAYERS * BATCH * HID * 2);
  unsigned short* hact   = (unsigned short*)alloc((size_t)BATCH * SEQ * HID * 2);
  float*          cstate = (float*)alloc((size_t)LAYERS * BATCH * HID * 4);
  float*          gates  = (float*)alloc((size_t)BATCH * 4 * HID * 4);
  float*          rowloss= (float*)alloc((size_t)BATCH * SEQ * 4);

  // init + precision conversion (+ weight transposes, once per call)
  zero_state_kernel<<<(LAYERS * BATCH * HID + 255) / 256, 256, 0, stream>>>(
      h_bf16, cstate, LAYERS * BATCH * HID);
  embed_kernel<<<2048, 256, 0, stream>>>(input_data, embedding, x_bf16);
  for (int l = 0; l < LAYERS; ++l)
    transpose_bf16_kernel<<<dim3(4 * HID / 32, 2 * HID / 32), dim3(32, 8), 0,
                            stream>>>(
        lstm_w + (size_t)l * 2 * HID * 4 * HID,
        W_t + (size_t)l * 4 * HID * 2 * HID, 2 * HID, 4 * HID);
  transpose_bf16_kernel<<<dim3(VOCAB / 32, HID / 32), dim3(32, 8), 0, stream>>>(
      softmax_w, Wsm_t, HID, VOCAB);

  // sequential LSTM
  for (int t = 0; t < SEQ; ++t) {
    for (int l = 0; l < LAYERS; ++l) {
      const unsigned short* Ain;
      int lda;
      if (l == 0) { Ain = x_bf16 + (size_t)t * HID; lda = SEQ * HID; }
      else        { Ain = h_bf16 + (size_t)(l - 1) * BATCH * HID; lda = HID; }
      lstm_gates_kernel<<<128, 64, 0, stream>>>(
          Ain, lda,
          h_bf16 + (size_t)l * BATCH * HID,
          W_t + (size_t)l * 4 * HID * 2 * HID,
          lstm_b + (size_t)l * 4 * HID,
          gates);
      lstm_point_kernel<<<(BATCH * HID + 255) / 256, 256, 0, stream>>>(
          gates,
          cstate + (size_t)l * BATCH * HID,
          h_bf16 + (size_t)l * BATCH * HID,
          hact, (l == LAYERS - 1) ? 1 : 0, t);
    }
  }

  // projection + loss
  logits_kernel<<<dim3(VOCAB / 128, (BATCH * SEQ) / 128), 256, 0, stream>>>(
      hact, Wsm_t, softmax_b, out);
  rowloss_kernel<<<BATCH * SEQ, 256, 0, stream>>>(out, targets, rowloss);
  cost_kernel<<<1, 256, 0, stream>>>(rowloss, out + (size_t)BATCH * SEQ * VOCAB);
}
